// ForceGNN_6536940224660
// MI455X (gfx1250) — matrix operand order repaired and checked by reading, not compiled, not run on hardware
//
#include <hip/hip_runtime.h>

// ---------------------------------------------------------------------------
// ForceGNN on MI455X (gfx1250, wave32, WMMA)
//   - message MLP (32x32) per 16-edge tile via v_wmma_f32_16x16x32_bf16
//   - 8 tiles per wave; edge indices (row/col) hoisted into a branch-free
//     prologue (register row offsets + one ds_store_b128 of dst offsets)
//   - double-buffered LDS A staging -> gather/WMMA overlap
//   - problem sizes divide exactly (static_assert) -> zero bounds logic
//   - node MLP (64x32 / 64x2) via 4 accumulating WMMAs; bf16 h copy fused
// ---------------------------------------------------------------------------

typedef __attribute__((ext_vector_type(16))) __bf16        v16bf;
typedef __attribute__((ext_vector_type(8)))  float         v8f;
typedef __attribute__((ext_vector_type(8)))  unsigned int  v8u;

union Frag16 { v8u u; v16bf h; };

#define GNN_N 100000
#define GNN_E 1600000
#define GNN_F 32
#define MSG_TPW 8       // edge tiles per wave (32 lanes x 4 = 128 edges/wave)
#define SA_PAD 40       // A tile row stride (halves): 80B, 16B-aligned, conflict-free
#define SA2_PAD 72      // node A tile row stride (halves)
#define SW_PAD 40       // msg weight [N][K] row stride
#define SW2_PAD 72      // node weight [N][K] row stride

static_assert(GNN_E % (16 * MSG_TPW * 4) == 0, "edge grid divides exactly");
static_assert(GNN_N % 16 == 0, "node tiles divide exactly");

// A-matrix 16x32 (16-bit) fragment dword v holds K = 2v + 16*half, +1 (same order as B)
static __device__ __forceinline__ int kbase_a(int v, int half) {
  return 2 * v + 16 * half;
}
// B-matrix 32x16 (16-bit) fragment dword v holds K = 2v + 16*half, +1
static __device__ __forceinline__ int kbase_b(int v, int half) {
  return 2 * v + 16 * half;
}

// ------------------------------ small kernels ------------------------------
__global__ void k_zero_f32(float* p, int n) {
  int i = blockIdx.x * blockDim.x + threadIdx.x;
  if (i < n) p[i] = 0.0f;
}

__global__ void k_count_deg(const long long* __restrict__ col, float* deg, int e_cnt) {
  int e = blockIdx.x * blockDim.x + threadIdx.x;
  if (e < e_cnt) atomicAdd(&deg[(int)col[e]], 1.0f);
}

__global__ void k_inv_deg(float* deg, int n) {
  int i = blockIdx.x * blockDim.x + threadIdx.x;
  if (i < n) deg[i] = 1.0f / fmaxf(deg[i], 1.0f);
}

__global__ void k_cast_bf16(const float* __restrict__ h, __bf16* __restrict__ h16, int n) {
  int i = blockIdx.x * blockDim.x + threadIdx.x;
  if (i < n) h16[i] = (__bf16)h[i];
}

// --------------------- edge kernel: message MLP + scatter -------------------
// Grid covers E exactly: no bounds checks anywhere.
__global__ __launch_bounds__(128) void k_msg_wmma(
    const __bf16* __restrict__ h16, const long long* __restrict__ row,
    const long long* __restrict__ col, const float* __restrict__ mw,
    const float* __restrict__ mb, float* __restrict__ agg) {
  __shared__ __align__(16) __bf16 sWt[32][SW_PAD];          // transposed [N][K]
  __shared__ float sBias[32];
  __shared__ __align__(16) __bf16 sA[4][2][16][SA_PAD];     // double-buffered A tiles
  __shared__ __align__(16) int sDst[4][16 * MSG_TPW];       // dst*32 for all 8 tiles

  const int tid  = threadIdx.x;
  const int wave = tid >> 5;
  const int lane = tid & 31;
  const int half = lane >> 4;
  const int idx  = lane & 15;

  // stage weights transposed (f32 -> bf16): sWt[n][k] = mw[k*32+n]
  for (int i = tid; i < 32 * 32; i += 128) {
    const int n = i & 31, k = i >> 5;
    sWt[n][k] = (__bf16)mw[k * 32 + n];
  }
  if (tid < 32) sBias[tid] = mb[tid];
  __syncthreads();

  const int tile0 = (blockIdx.x * 4 + wave) * MSG_TPW;
  const int ebase = tile0 * 16;

  // ---- branch-free index prologue -----------------------------------------
  // col: 4 consecutive int64 per lane via two b128 loads -> one ds_store_b128
  {
    const int4* c4 = (const int4*)(col + ebase + lane * 4);
    const int4 lo = c4[0], hi = c4[1];           // (lo.x,lo.z,hi.x,hi.z) = dsts
    int4 d;
    d.x = lo.x << 5;  // dst * GNN_F
    d.y = lo.z << 5;
    d.z = hi.x << 5;
    d.w = hi.z << 5;
    *(int4*)&sDst[wave][lane * 4] = d;
  }
  // row: this lane's 8 per-tile gather offsets, pre-loaded as one VMEM burst
  const int r  = lane >> 1;   // gather: 2 lanes per edge row
  const int hs = lane & 1;
  int rowoff[MSG_TPW];
#pragma unroll
  for (int t = 0; t < MSG_TPW; ++t)
    rowoff[t] = (int)row[ebase + t * 16 + r] * GNN_F;

  if (lane == 0 && tile0 + MSG_TPW < GNN_E / 16)
    __builtin_prefetch(&col[ebase + 16 * MSG_TPW], 0, 1);

  // B fragments: loop-invariant across all tiles of this wave
  Frag16 b0, b1;
#pragma unroll
  for (int v = 0; v < 8; ++v) {
    const int kb = kbase_b(v, half);
    b0.u[v] = *(const unsigned int*)&sWt[idx][kb];
    b1.u[v] = *(const unsigned int*)&sWt[idx + 16][kb];
  }
  const float bias0 = sBias[idx], bias1 = sBias[idx + 16];

  // ---- steady state: gather -> WMMA -> atomic scatter, 8 tiles ------------
#pragma unroll
  for (int t = 0; t < MSG_TPW; ++t) {
    const int buf = t & 1;

    // stage A tile (bf16 rows, 32B per lane)
    {
      const uint4* p = (const uint4*)(h16 + (size_t)rowoff[t] + hs * 16);
      uint4* q = (uint4*)&sA[wave][buf][r][hs * 16];
      q[0] = p[0];
      q[1] = p[1];
    }
    // same-wave LDS ops are in-order (DScnt); disjoint buffers let the
    // scheduler overlap this stage with the previous tile's WMMA/scatter

    // A fragment: contiguous K pairs -> two ds_load_b128 per lane
    Frag16 a;
#pragma unroll
    for (int v = 0; v < 8; ++v)
      a.u[v] = *(const unsigned int*)&sA[wave][buf][idx][kbase_a(v, half)];

    v8f c0 = {}, c1 = {};
    c0 = __builtin_amdgcn_wmma_f32_16x16x32_bf16(false, a.h, false, b0.h, (short)0, c0, false, false);
    c1 = __builtin_amdgcn_wmma_f32_16x16x32_bf16(false, a.h, false, b1.h, (short)0, c1, false, false);

    // bias + ReLU, branch-free atomic scatter (C: M = rr + 8*half, N = idx)
#pragma unroll
    for (int rr = 0; rr < 8; ++rr) {
      const float m0 = fmaxf(c0[rr] + bias0, 0.0f);
      const float m1 = fmaxf(c1[rr] + bias1, 0.0f);
      const int dstoff = sDst[wave][t * 16 + 8 * half + rr];
      atomicAdd(&agg[dstoff + idx], m0);
      atomicAdd(&agg[dstoff + 16 + idx], m1);
    }
  }
}

// ------------------- node kernel: [h | agg/deg] @ W + b ---------------------
// All active tiles are full (N % 16 == 0); writes f32 h and fused bf16 copy.
__global__ __launch_bounds__(128) void k_node_wmma(
    const float* __restrict__ h_in, const float* __restrict__ agg,
    const float* __restrict__ inv_deg, const float* __restrict__ w,
    const float* __restrict__ b, const float* __restrict__ resid,
    float* __restrict__ out, __bf16* __restrict__ h16_out,
    int nout, int do_relu, int n_cnt) {
  __shared__ __align__(16) __bf16 sWt[32][SW2_PAD];         // transposed [N][K=0..63]
  __shared__ float sBias[32];
  __shared__ __align__(16) __bf16 sA[4][16][SA2_PAD];

  const int tid  = threadIdx.x;
  const int wave = tid >> 5;
  const int lane = tid & 31;
  const int half = lane >> 4;
  const int idx  = lane & 15;

  for (int i = tid; i < 64 * 32; i += 128) {
    const int n = i & 31, k = i >> 5;
    sWt[n][k] = (n < nout) ? (__bf16)w[k * nout + n] : (__bf16)0.0f;
  }
  if (tid < 32) sBias[tid] = (tid < nout) ? b[tid] : 0.0f;

  const int n0     = (blockIdx.x * 4 + wave) * 16;
  const bool active = (n0 < n_cnt);

  if (active) {
    const int r    = lane >> 1;
    const int hs   = lane & 1;  // 0: h part (K 0..31), 1: agg/deg part (K 32..63)
    const int node = n0 + r;
    if (hs == 0) {
      const float4* p = (const float4*)(h_in + (size_t)node * GNN_F);
#pragma unroll
      for (int t = 0; t < 8; ++t) {
        float4 v = p[t];
        sA[wave][r][4 * t + 0] = (__bf16)v.x;
        sA[wave][r][4 * t + 1] = (__bf16)v.y;
        sA[wave][r][4 * t + 2] = (__bf16)v.z;
        sA[wave][r][4 * t + 3] = (__bf16)v.w;
      }
    } else {
      const float sc = inv_deg[node];
      const float4* p = (const float4*)(agg + (size_t)node * GNN_F);
#pragma unroll
      for (int t = 0; t < 8; ++t) {
        float4 v = p[t];
        sA[wave][r][32 + 4 * t + 0] = (__bf16)(v.x * sc);
        sA[wave][r][32 + 4 * t + 1] = (__bf16)(v.y * sc);
        sA[wave][r][32 + 4 * t + 2] = (__bf16)(v.z * sc);
        sA[wave][r][32 + 4 * t + 3] = (__bf16)(v.w * sc);
      }
    }
  }
  __syncthreads();
  if (!active) return;

  Frag16 b00, b01, b10, b11, a0, a1;
#pragma unroll
  for (int v = 0; v < 8; ++v) {
    const int kb = kbase_b(v, half);
    const int ka = kbase_a(v, half);
    b00.u[v] = *(const unsigned int*)&sWt[idx][kb];
    b01.u[v] = *(const unsigned int*)&sWt[idx + 16][kb];
    b10.u[v] = *(const unsigned int*)&sWt[idx][32 + kb];
    b11.u[v] = *(const unsigned int*)&sWt[idx + 16][32 + kb];
    a0.u[v]  = *(const unsigned int*)&sA[wave][idx][ka];
    a1.u[v]  = *(const unsigned int*)&sA[wave][idx][32 + ka];
  }

  v8f c0 = {}, c1 = {};
  c0 = __builtin_amdgcn_wmma_f32_16x16x32_bf16(false, a0.h, false, b00.h, (short)0, c0, false, false);
  c0 = __builtin_amdgcn_wmma_f32_16x16x32_bf16(false, a1.h, false, b10.h, (short)0, c0, false, false);
  c1 = __builtin_amdgcn_wmma_f32_16x16x32_bf16(false, a0.h, false, b01.h, (short)0, c1, false, false);
  c1 = __builtin_amdgcn_wmma_f32_16x16x32_bf16(false, a1.h, false, b11.h, (short)0, c1, false, false);

  const float bias0 = sBias[idx], bias1 = sBias[idx + 16];
#pragma unroll
  for (int r = 0; r < 8; ++r) {
    const int node = n0 + 8 * half + r;
    float v0 = c0[r] + bias0;
    float v1 = c1[r] + bias1;
    if (do_relu) {
      v0 = fmaxf(v0, 0.0f);
      v1 = fmaxf(v1, 0.0f);
    }
    if (resid) {  // reference: h = relu(out) + h_prev
      v0 += resid[(size_t)node * GNN_F + idx];
      v1 += resid[(size_t)node * GNN_F + 16 + idx];
    }
    if (idx < nout)      out[(size_t)node * nout + idx] = v0;
    if (idx + 16 < nout) out[(size_t)node * nout + 16 + idx] = v1;
    if (h16_out) {       // fused bf16 copy for the next layer's edge gather
      h16_out[(size_t)node * GNN_F + idx] = (__bf16)v0;
      h16_out[(size_t)node * GNN_F + 16 + idx] = (__bf16)v1;
    }
  }
}

// ------------------------------- launcher ----------------------------------
extern "C" void kernel_launch(void* const* d_in, const int* in_sizes, int n_in,
                              void* d_out, int out_size, void* d_ws, size_t ws_size,
                              hipStream_t stream) {
  const float*     x      = (const float*)d_in[0];
  const long long* ei     = (const long long*)d_in[1];  // [2, E] int64
  // d_in[2] = batch (unused by the reference computation)
  const float*     msg_w  = (const float*)d_in[3];  // [4,32,32]
  const float*     msg_b  = (const float*)d_in[4];  // [4,32]
  const float*     out_w  = (const float*)d_in[5];  // [3,64,32]
  const float*     out_b  = (const float*)d_in[6];  // [3,32]
  const float*     last_w = (const float*)d_in[7];  // [64,2]
  const float*     last_b = (const float*)d_in[8];  // [2]
  float*           out    = (float*)d_out;

  const long long* row = ei;
  const long long* col = ei + GNN_E;

  // workspace carve-out (256B aligned slices)
  char*  ws  = (char*)d_ws;
  size_t off = 0;
  auto carve = [&](size_t bytes) -> void* {
    void* p = ws + off;
    off += (bytes + 255) & ~(size_t)255;
    return p;
  };
  float*  hA  = (float*)carve((size_t)GNN_N * GNN_F * sizeof(float));
  float*  hB  = (float*)carve((size_t)GNN_N * GNN_F * sizeof(float));
  float*  agg = (float*)carve((size_t)GNN_N * GNN_F * sizeof(float));
  float*  deg = (float*)carve((size_t)GNN_N * sizeof(float));
  __bf16* h16 = (__bf16*)carve((size_t)GNN_N * GNN_F * sizeof(__bf16));

  const int nf      = GNN_N * GNN_F;
  const int blkN    = (GNN_N + 255) / 256;
  const int blkNF   = (nf + 255) / 256;
  const int blkE    = (GNN_E + 255) / 256;
  const int msgBlks = GNN_E / (16 * MSG_TPW * 4);      // exact
  const int nodBlks = (GNN_N / 16 + 3) / 4;

  // degree (layer-invariant): deg -> 1/max(deg,1)
  k_zero_f32<<<blkN, 256, 0, stream>>>(deg, GNN_N);
  k_count_deg<<<blkE, 256, 0, stream>>>(col, deg, GNN_E);
  k_inv_deg<<<blkN, 256, 0, stream>>>(deg, GNN_N);

  // bf16 copy of layer-0 input; later layers fuse it into the node kernel
  k_cast_bf16<<<blkNF, 256, 0, stream>>>(x, h16, nf);

  const float* hin = x;
  float* pingpong[2] = {hA, hB};
  for (int l = 0; l < 4; ++l) {
    k_zero_f32<<<blkNF, 256, 0, stream>>>(agg, nf);
    k_msg_wmma<<<msgBlks, 128, 0, stream>>>(h16, row, col,
                                            msg_w + (size_t)l * 32 * 32,
                                            msg_b + (size_t)l * 32, agg);
    if (l < 3) {
      float* hout = pingpong[l & 1];              // l0->hA, l1->hB, l2->hA
      const float* resid = (l == 0) ? nullptr : hin;
      k_node_wmma<<<nodBlks, 128, 0, stream>>>(hin, agg, deg,
                                               out_w + (size_t)l * 64 * 32,
                                               out_b + (size_t)l * 32, resid,
                                               hout, h16,
                                               /*nout=*/32, /*relu=*/1, GNN_N);
      hin = hout;
    } else {
      k_node_wmma<<<nodBlks, 128, 0, stream>>>(hin, agg, deg, last_w, last_b,
                                               nullptr, out, nullptr,
                                               /*nout=*/2, /*relu=*/0, GNN_N);
    }
  }
}
